// AE_spikes_60902636257522
// MI455X (gfx1250) — matrix operand order repaired and checked by reading, not compiled, not run on hardware
//
#include <hip/hip_runtime.h>

// ---------------------------------------------------------------------------
// CDNA5 (gfx1250, wave32) spiking-autoencoder forward.
//   1) quantize features once (digitize -> /16 -> clip) to f16, padded K=800
//   2) convert weights to f16 (W0 padded to K=800)
//   3) one WMMA GEMM for Z0 = qf @ W0^T  (constant-over-time layer-0 drive)
//   4) fused recurrent kernel: 16 timesteps, LIF state in VGPR C-fragments,
//      W1/W2 B-fragments register-resident, W3 staged once into LDS via
//      GLOBAL_LOAD_ASYNC_TO_LDS_B128 (bank-conflict-padded rows), all matmuls
//      via v_wmma_f32_16x16x32_f16.
// ---------------------------------------------------------------------------

typedef __attribute__((ext_vector_type(16))) _Float16 v16h;
typedef __attribute__((ext_vector_type(8)))  float    v8f;
typedef int i32x4 __attribute__((__vector_size__(16)));

#define AS1 __attribute__((address_space(1)))
#define AS3 __attribute__((address_space(3)))

#define ROWP 136  // padded LDS row stride in halves: 272B = 68 banks (gcd-free)

union AF { v16h v; float4 q[2]; };

// A fragment (16x32 f16, M x K): lane l: m = l&15, g = l>>4.
// halves 0..7  -> K = kb*32 + g*8 + h ; halves 8..15 -> K = kb*32 + 16 + g*8
__device__ __forceinline__ v16h load_frag_a(const _Float16* base, int ld,
                                            int m, int g, int kb) {
  AF f;
  const _Float16* p = base + m * ld + kb * 32 + g * 8;
  f.q[0] = *reinterpret_cast<const float4*>(p);
  f.q[1] = *reinterpret_cast<const float4*>(p + 16);
  return f.v;
}

// B fragment (32x16 f16, K x N) from row-major W[n][k]:
// lane l: n = l&15, g = l>>4; halves 0..7 -> K = kb*32 + g*8 + h ; halves 8..15 -> K = kb*32 + 16 + g*8
__device__ __forceinline__ v16h load_frag_b(const _Float16* base, int ld,
                                            int nrow, int g, int kb) {
  AF f;
  const _Float16* p = base + nrow * ld + kb * 32 + g * 8;
  f.q[0] = *reinterpret_cast<const float4*>(p);
  f.q[1] = *reinterpret_cast<const float4*>(p + 16);
  return f.v;
}

__device__ __forceinline__ v8f wmma_f16(v16h a, v16h b, v8f c) {
  return __builtin_amdgcn_wmma_f32_16x16x32_f16(
      false, a, false, b, (short)0, c, false, false);
}

// LIF update over a 16x16 C-fragment slice (layers 0..2 emit clipped spikes
// to LDS with the pre-decrement rule; layer 3 passes xdst=nullptr).
__device__ __forceinline__ void lif_step(v8f& t, v8f& s, float bb,
                                         _Float16* xdst, int g, int col) {
#pragma unroll
  for (int v = 0; v < 8; ++v) {
    float tt = t[v];
    float fired = (tt >= 1.0f) ? 1.0f : 0.0f;
    tt -= fired;
    float o = (tt == bb) ? 0.0f : fired;   // output[totals == bias] = 0
    tt = (o > 0.0f) ? 0.0f : tt;           // totals[output > 0] = 0
    t[v] = tt;
    float sn = s[v] + o;                   // spike_layers += output
    if (xdst) {
      xdst[(v + g * 8) * ROWP + col] = (_Float16)fminf(sn, 1.0f); // clip
      s[v] = (sn > 0.0f) ? sn - 1.0f : sn; // input_spikes[>0] -= 1
    } else {
      s[v] = sn;
    }
  }
}

// --- prep: quantize features to f16, pad K 784 -> 800 with zeros -----------
__global__ void k_quant(const float* __restrict__ f, _Float16* __restrict__ qf) {
  int idx = blockIdx.x * blockDim.x + threadIdx.x;
  if (idx >= 8192 * 800) return;
  int r = idx / 800, c = idx - r * 800;
  float q = 0.0f;
  if (c < 784) {
    float x = f[r * 784 + c];
    q = floorf(x * 16.0f) * 0.0625f;       // (digitize - 1) / 16
    q = fminf(fmaxf(q, 0.0f), 1.0f);       // clip
  }
  qf[idx] = (_Float16)q;
}

// --- prep: f32 -> f16 weight conversion with optional column padding -------
__global__ void k_cvt(const float* __restrict__ src, _Float16* __restrict__ dst,
                      int rows, int cols, int ldp) {
  int idx = blockIdx.x * blockDim.x + threadIdx.x;
  if (idx >= rows * ldp) return;
  int r = idx / ldp, c = idx - r * ldp;
  dst[idx] = (_Float16)((c < cols) ? src[r * cols + c] : 0.0f);
}

// --- Z0 = qf @ W0^T : [8192,800] x [128,800]^T -> [8192,128] f32 -----------
__global__ __launch_bounds__(256) void k_gemm_z0(const _Float16* __restrict__ qf,
                                                 const _Float16* __restrict__ w0h,
                                                 float* __restrict__ z0) {
  const int tid = threadIdx.x;
  const int w = tid >> 5, l = tid & 31;
  const int g = l >> 4, n = l & 15, m = l & 15;
  const int rowbase = blockIdx.x * 16;
  v8f acc;
#pragma unroll
  for (int v = 0; v < 8; ++v) acc[v] = 0.0f;
#pragma unroll
  for (int kb = 0; kb < 25; ++kb) {
    v16h a = load_frag_a(qf + (size_t)rowbase * 800, 800, m, g, kb);
    v16h b = load_frag_b(w0h, 800, w * 16 + n, g, kb);
    acc = wmma_f16(a, b, acc);
  }
#pragma unroll
  for (int v = 0; v < 8; ++v)
    z0[(size_t)(rowbase + v + g * 8) * 128 + w * 16 + n] = acc[v];
}

// --- fused 16-timestep recurrent LIF network -------------------------------
__global__ __launch_bounds__(256) void k_snn(
    const float* __restrict__ z0,
    const _Float16* __restrict__ w1h, const _Float16* __restrict__ w2h,
    const _Float16* __restrict__ w3h,
    const float* __restrict__ b0, const float* __restrict__ b1,
    const float* __restrict__ b2, const float* __restrict__ b3,
    float* __restrict__ out) {
  // W3 staged once (784 rows x 128 halves, rows padded to ROWP): ~208 KB.
  __shared__ __align__(16) _Float16 w3s[784 * ROWP];
  __shared__ __align__(16) _Float16 xb[3][16 * ROWP];  // inter-layer spikes

  const int tid = threadIdx.x;
  const int w = tid >> 5, l = tid & 31;      // wave 0..7, lane 0..31 (wave32)
  const int g = l >> 4, n = l & 15, m = l & 15;
  const int rowbase = blockIdx.x * 16;
  const int col = w * 16 + n;                // this wave's hidden-col slice

  // ---- stage W3 into LDS (async-to-LDS path if the toolchain exposes it) --
  for (int i = tid; i < 784 * 8; i += 256) {         // 16B chunks
    int row = i >> 3, ch = i & 7;
    const _Float16* src = w3h + row * 128 + ch * 8;
    _Float16* dst = &w3s[row * ROWP + ch * 8];
#if __has_builtin(__builtin_amdgcn_global_load_async_to_lds_b128)
    __builtin_amdgcn_global_load_async_to_lds_b128(
        (AS1 i32x4*)(AS1 void*)const_cast<_Float16*>(src),
        (AS3 i32x4*)(AS3 void*)dst, 0, 0);
#else
    *reinterpret_cast<float4*>(dst) = *reinterpret_cast<const float4*>(src);
#endif
  }
#if __has_builtin(__builtin_amdgcn_global_load_async_to_lds_b128)
#if __has_builtin(__builtin_amdgcn_s_wait_asynccnt)
  __builtin_amdgcn_s_wait_asynccnt(0);
#else
  asm volatile("s_wait_asynccnt 0" ::: "memory");
#endif
#endif

  const float bb0 = b0[col], bb1 = b1[col], bb2 = b2[col];

  // Preload W1/W2 B-fragments (this wave's N-tile) into registers.
  v16h w1f[4], w2f[4];
#pragma unroll
  for (int kb = 0; kb < 4; ++kb) {
    w1f[kb] = load_frag_b(w1h, 128, col, g, kb);
    w2f[kb] = load_frag_b(w2h, 128, col, g, kb);
  }

  // Constant layer-0 drive fragment.
  v8f z0f;
#pragma unroll
  for (int v = 0; v < 8; ++v)
    z0f[v] = z0[(size_t)(rowbase + v + g * 8) * 128 + col];

  // State: totals init to bias (reset_totals), spike counters zero.
  v8f t0, t1, t2, s0, s1, s2;
#pragma unroll
  for (int v = 0; v < 8; ++v) {
    t0[v] = bb0; t1[v] = bb1; t2[v] = bb2;
    s0[v] = 0.0f; s1[v] = 0.0f; s2[v] = 0.0f;
  }

  // Layer 3: 49 N-tiles striped over 8 waves (wave w: tiles w, w+8, ...).
  v8f t3[7], s3[7];
  float bb3[7];
#pragma unroll
  for (int j = 0; j < 7; ++j) {
    int tile = w + 8 * j;
    bb3[j] = (tile < 49) ? b3[tile * 16 + n] : 0.0f;
#pragma unroll
    for (int v = 0; v < 8; ++v) { t3[j][v] = bb3[j]; s3[j][v] = 0.0f; }
  }

  __syncthreads();  // W3 staged, xb clean

#pragma unroll 1
  for (int step = 0; step < 16; ++step) {
    // layer 0: constant drive, elementwise LIF, emit x1 to LDS
    t0 = t0 + z0f;
    lif_step(t0, s0, bb0, xb[0], g, col);
    __syncthreads();

    // layer 1: t1 += x1 @ W1^T (wave-private N-tile), LIF, emit x2
#pragma unroll
    for (int kb = 0; kb < 4; ++kb)
      t1 = wmma_f16(load_frag_a(xb[0], ROWP, m, g, kb), w1f[kb], t1);
    lif_step(t1, s1, bb1, xb[1], g, col);
    __syncthreads();

    // layer 2
#pragma unroll
    for (int kb = 0; kb < 4; ++kb)
      t2 = wmma_f16(load_frag_a(xb[1], ROWP, m, g, kb), w2f[kb], t2);
    lif_step(t2, s2, bb2, xb[2], g, col);
    __syncthreads();

    // layer 3: shared A-fragments, per-tile B served from LDS
    v16h a3[4];
#pragma unroll
    for (int kb = 0; kb < 4; ++kb) a3[kb] = load_frag_a(xb[2], ROWP, m, g, kb);
#pragma unroll
    for (int j = 0; j < 7; ++j) {
      int tile = w + 8 * j;
      if (tile < 49) {
#pragma unroll
        for (int kb = 0; kb < 4; ++kb)
          t3[j] = wmma_f16(a3[kb],
                           load_frag_b(w3s, ROWP, tile * 16 + n, g, kb), t3[j]);
        lif_step(t3[j], s3[j], bb3[j], nullptr, g, col);
      }
    }
    __syncthreads();
  }

  // reconstructed = spike_layers[3] * (1/16)
#pragma unroll
  for (int j = 0; j < 7; ++j) {
    int tile = w + 8 * j;
    if (tile < 49) {
#pragma unroll
      for (int v = 0; v < 8; ++v)
        out[(size_t)(rowbase + v + g * 8) * 784 + tile * 16 + n] =
            s3[j][v] * 0.0625f;
    }
  }
}

// ---------------------------------------------------------------------------
extern "C" void kernel_launch(void* const* d_in, const int* in_sizes, int n_in,
                              void* d_out, int out_size, void* d_ws, size_t ws_size,
                              hipStream_t stream) {
  (void)in_sizes; (void)n_in; (void)out_size; (void)ws_size;
  const float* feat = (const float*)d_in[0];
  const float* W0 = (const float*)d_in[1];
  const float* b0 = (const float*)d_in[2];
  const float* W1 = (const float*)d_in[3];
  const float* b1 = (const float*)d_in[4];
  const float* W2 = (const float*)d_in[5];
  const float* b2 = (const float*)d_in[6];
  const float* W3 = (const float*)d_in[7];
  const float* b3 = (const float*)d_in[8];

  char* ws = (char*)d_ws;
  size_t off = 0;
  auto alloc = [&](size_t bytes) {
    void* p = ws + off;
    off += (bytes + 255) & ~size_t(255);
    return p;
  };
  _Float16* qf  = (_Float16*)alloc((size_t)8192 * 800 * 2);  // ~12.8 MB
  _Float16* w0h = (_Float16*)alloc((size_t)128 * 800 * 2);
  _Float16* w1h = (_Float16*)alloc((size_t)128 * 128 * 2);
  _Float16* w2h = (_Float16*)alloc((size_t)128 * 128 * 2);
  _Float16* w3h = (_Float16*)alloc((size_t)784 * 128 * 2);
  float*    z0  = (float*)   alloc((size_t)8192 * 128 * 4);  // ~4.2 MB

  k_quant<<<(8192 * 800 + 255) / 256, 256, 0, stream>>>(feat, qf);
  k_cvt<<<(128 * 800 + 255) / 256, 256, 0, stream>>>(W0, w0h, 128, 784, 800);
  k_cvt<<<(128 * 128 + 255) / 256, 256, 0, stream>>>(W1, w1h, 128, 128, 128);
  k_cvt<<<(128 * 128 + 255) / 256, 256, 0, stream>>>(W2, w2h, 128, 128, 128);
  k_cvt<<<(784 * 128 + 255) / 256, 256, 0, stream>>>(W3, w3h, 784, 128, 128);

  k_gemm_z0<<<512, 256, 0, stream>>>(qf, w0h, z0);
  k_snn<<<512, 256, 0, stream>>>(z0, w1h, w2h, w3h, b0, b1, b2, b3,
                                 (float*)d_out);
}
